// MaximalCliqueSearcher_80436147519553
// MI455X (gfx1250) — matrix operand order repaired and checked by reading, not compiled, not run on hardware
//
#include <hip/hip_runtime.h>
#include <hip/hip_bf16.h>
#include <math.h>

// ---------------------------------------------------------------------------
// MaximalCliqueSearcher on MI455X (gfx1250, wave32, WMMA)
//
// Pipeline:
//  K1 score_kernel   : points -> score0 (f32 + f16)           [384x384]
//  K2 wmma_symsq     : sc2 = score0 @ score0   (v_wmma f16)   [384x384 f32]
//  K3 combine_kernel : scoreF = score0*sc2 ; T = cbrt(offdiag) (f32+f16)
//  K4 wmma_symsq     : T2 = T @ T              (v_wmma f16)
//  K5 rowstats       : wijk = .5*sum_j T*T2 ; degree ; cf ; sums
//  K6 sortdesc       : cf_sorted (rank sort, 1 block)
//  K7 otsu_scalars   : OTSU thresh + f*tmp scalars (1 thread, serial, det.)
//  K8 finalize       : out = triu(prune(scoreF), 1)
//
// The O(M^3) cbrt broadcast collapses to diag(T^3) = rowdot(T, T@T) since
// T = cbrt(S) elementwise for S >= 0 and T is symmetric with zero diagonal.
// ---------------------------------------------------------------------------

#define M 384
#define MM (M * M)

typedef __attribute__((ext_vector_type(16))) _Float16 v16h;
typedef __attribute__((ext_vector_type(8)))  _Float16 v8h;
typedef __attribute__((ext_vector_type(8)))  float    v8f;

// ---- K1: pairwise distance-difference score --------------------------------
__global__ void score_kernel(const float* __restrict__ P,
                             float* __restrict__ s32,
                             _Float16* __restrict__ s16) {
    int idx = blockIdx.x * blockDim.x + threadIdx.x;
    if (idx >= MM) return;
    int i = idx / M, j = idx % M;
    const float* pi = P + i * 6;
    const float* pj = P + j * 6;
    float dx = pi[0] - pj[0], dy = pi[1] - pj[1], dz = pi[2] - pj[2];
    float sq1 = dx * dx + dy * dy + dz * dz;
    float d1 = (sq1 > 0.f) ? sqrtf(sq1) : 0.f;
    dx = pi[3] - pj[3]; dy = pi[4] - pj[4]; dz = pi[5] - pj[5];
    float sq2 = dx * dx + dy * dy + dz * dz;
    float d2 = (sq2 > 0.f) ? sqrtf(sq2) : 0.f;
    float d = fabsf(d1 - d2);
    float sc = __expf(-d * d * (1.0f / (2.0f * 0.3f * 0.3f)));
    if (sc < 0.5f) sc = 0.f;
    s32[idx] = sc;
    s16[idx] = (_Float16)sc;
}

// ---- K2/K4: C = A @ A for symmetric A (f16 in, f32 out) via WMMA -----------
// Register-blocked: each wave computes a 16(M) x 64(N) strip = 4 WMMA tiles,
// reusing one A fragment across 4 B fragments per K-step.
// Tiles: 24 (tm) x 6 (tn-group) = 144 waves; 4 waves/block -> 36 blocks.
__global__ void wmma_symsq(const _Float16* __restrict__ A,
                           float* __restrict__ C) {
    int wave = threadIdx.x >> 5;
    int lane = threadIdx.x & 31;
    int strip = blockIdx.x * 4 + wave;           // 0..143
    int tm = strip / (M / 64);                   // 0..23 (row tile)
    int tg = strip % (M / 64);                   // 0..5  (group of 4 col tiles)
    int half = lane >> 4;                        // lane group 0/1
    int l16  = lane & 15;

    const _Float16* rowA = A + (tm * 16 + l16) * M;          // A frag: M = l16
    // B = A^T = A (symmetric): fragment t reads row N = tg*64 + t*16 + l16
    const _Float16* rowB0 = A + (tg * 64 +  0 + l16) * M;
    const _Float16* rowB1 = A + (tg * 64 + 16 + l16) * M;
    const _Float16* rowB2 = A + (tg * 64 + 32 + l16) * M;
    const _Float16* rowB3 = A + (tg * 64 + 48 + l16) * M;

    v8f acc0 = {}, acc1 = {}, acc2 = {}, acc3 = {};
    #pragma unroll 2
    for (int k0 = 0; k0 < M; k0 += 32) {
        // 16-bit A layout: lane half h -> elems 0..15 = K[k0+16h ..] (contig)
        v16h a = *(const v16h*)(rowA + k0 + half * 16);
        // 16-bit B layout: lane half h -> elems 0..15 = K[k0+16h ..] (contig)
        v16h b0 = *(const v16h*)(rowB0 + k0 + half * 16);
        v16h b1 = *(const v16h*)(rowB1 + k0 + half * 16);
        v16h b2 = *(const v16h*)(rowB2 + k0 + half * 16);
        v16h b3 = *(const v16h*)(rowB3 + k0 + half * 16);
        acc0 = __builtin_amdgcn_wmma_f32_16x16x32_f16(false, a, false, b0,
                                                      (short)0, acc0, false, false);
        acc1 = __builtin_amdgcn_wmma_f32_16x16x32_f16(false, a, false, b1,
                                                      (short)0, acc1, false, false);
        acc2 = __builtin_amdgcn_wmma_f32_16x16x32_f16(false, a, false, b2,
                                                      (short)0, acc2, false, false);
        acc3 = __builtin_amdgcn_wmma_f32_16x16x32_f16(false, a, false, b3,
                                                      (short)0, acc3, false, false);
        // hint next K-block of the shared A row into cache (global_prefetch_b8)
        if (k0 + 32 < M) __builtin_prefetch(rowA + k0 + 32, 0, 1);
    }
    // C/D layout: VGPR v -> M = half*8 + v ; N = l16
    float* crow = C + (tm * 16 + half * 8) * M + tg * 64 + l16;
    #pragma unroll
    for (int v = 0; v < 8; ++v) {
        crow[v * M +  0] = acc0[v];
        crow[v * M + 16] = acc1[v];
        crow[v * M + 32] = acc2[v];
        crow[v * M + 48] = acc3[v];
    }
}

// ---- K3: scoreF = score0*sc2 (in place into sc2); T = cbrt(off-diag) -------
__global__ void combine_kernel(const float* __restrict__ s0,
                               float* __restrict__ scoreF,      // holds sc2 on entry
                               float* __restrict__ Tf,
                               _Float16* __restrict__ Th) {
    int idx = blockIdx.x * blockDim.x + threadIdx.x;
    if (idx >= MM) return;
    int i = idx / M, j = idx % M;
    float s = s0[idx] * scoreF[idx];
    scoreF[idx] = s;
    float sz = (i == j) ? 0.f : s;
    float t = (sz > 0.f) ? cbrtf(sz) : 0.f;
    Tf[idx] = t;
    Th[idx] = (_Float16)t;
}

// ---- K5: per-row stats + deterministic tree-reduced sums -------------------
// wijk[i] = 0.5 * sum_j T[i,j]*T2[i,j] ; degree[i] = #nonzeros of row i
// scal[0] = total_factor ; scal[1] = sum(cf)
__global__ void rowstats_kernel(const float* __restrict__ Tf,
                                const float* __restrict__ T2,
                                float* __restrict__ cf,
                                float* __restrict__ scal) {
    __shared__ float s1[512], s2[512], s3[512];
    int i = threadIdx.x;
    float f1 = 0.f, f2 = 0.f, cfi = 0.f;
    if (i < M) {
        float wijk = 0.f;
        int deg = 0;
        const float* tr  = Tf + i * M;
        const float* t2r = T2 + i * M;
        for (int j = 0; j < M; ++j) {
            float t = tr[j];
            wijk += t * t2r[j];
            deg  += (t != 0.f) ? 1 : 0;
        }
        wijk *= 0.5f;
        bool invalid = (deg <= 1);
        float d = invalid ? 0.f : (float)deg;
        f1 = invalid ? 0.f : wijk;
        f2 = d * (d - 1.f) * 0.5f;
        cfi = f1 / (invalid ? 1.f : f2);
        cf[i] = cfi;
    }
    s1[i] = f1; s2[i] = f2; s3[i] = cfi;
    __syncthreads();
    for (int st = 256; st > 0; st >>= 1) {
        if (i < st) { s1[i] += s1[i + st]; s2[i] += s2[i + st]; s3[i] += s3[i + st]; }
        __syncthreads();
    }
    if (i == 0) {
        scal[0] = s1[0] / (s2[0] + 1e-10f);   // total_factor
        scal[1] = s3[0];                       // sum(cf)
    }
}

// ---- K6: descending rank sort of 384 values (one block) --------------------
__global__ void sortdesc_kernel(const float* __restrict__ cf,
                                float* __restrict__ out) {
    __shared__ float v[M];
    int i = threadIdx.x;
    v[i] = cf[i];
    __syncthreads();
    float x = v[i];
    int r = 0;
    for (int j = 0; j < M; ++j) {
        float y = v[j];
        r += (y > x || (y == x && j < i)) ? 1 : 0;
    }
    out[r] = x;
}

// ---- K7: OTSU threshold + prune scalars (single thread, fully serial) ------
// scal[2] = thresh ; scal[3] = f * tmp
__global__ void otsu_kernel(const float* __restrict__ cs,   // cf_sorted (desc)
                            float* __restrict__ scal) {
    if (threadIdx.x != 0) return;
    const float sum = scal[1];
    const float total_factor = scal[0];
    const float vmax = cs[0];
    const float vmin = cs[M - 1];
    float step = (vmax - vmin) * 0.01f;        // /QUANT
    float stepd = (step == 0.f) ? 1.f : step;
    float hist[100], shist[100];
    for (int b = 0; b < 100; ++b) { hist[b] = 0.f; shist[b] = 0.f; }
    for (int n = 0; n < M; ++n) {
        float val = cs[n];
        int id = (int)(val / stepd);           // trunc toward 0, like .to(int32)
        if (id >= 0) {
            if (id > 99) id = 99;
            hist[id] += 1.f;
            shist[id] += val;
        }
    }
    float n1 = 0.f, fore = 0.f;
    float best = -INFINITY; int bi = 0;
    for (int b = 0; b < 100; ++b) {
        n1 += hist[b]; fore += shist[b];
        float n2 = (float)M - n1;
        bool ok = (n1 > 0.f) && (n2 > 0.f);
        float m1 = fore / ((n1 > 0.f) ? n1 : 1.f);
        float m2 = (sum - fore) / ((n2 > 0.f) ? n2 : 1.f);
        float diff = m1 - m2;
        float sb = ok ? (n1 * n2 * diff * diff) : -INFINITY;
        if (sb > best) { best = sb; bi = b; }  // first strict max
    }
    float otsu = (float)bi * (vmax - vmin) * 0.01f;
    if (cs[0] == 0.f) otsu = 0.f;
    float avg = sum * (1.0f / (float)M);
    float thresh = fminf(otsu, fminf(avg, total_factor));
    float tmp = fmaxf(otsu, total_factor);
    float c49 = cs[49];
    float k = floorf((10.f * tmp - c49) / (0.05f * ((tmp == 0.f) ? 1.f : tmp))) + 1.f;
    float f = (10.f * tmp > c49) ? (10.f - 0.05f * k) : 10.f;
    scal[2] = thresh;
    scal[3] = f * tmp;
}

// ---- K8: prune + upper triangle -------------------------------------------
__global__ void finalize_kernel(const float* __restrict__ scoreF,
                                const float* __restrict__ cf,
                                const float* __restrict__ scal,
                                float* __restrict__ out) {
    int idx = blockIdx.x * blockDim.x + threadIdx.x;
    if (idx >= MM) return;
    int i = idx / M, j = idx % M;
    float thresh = scal[2];
    float ftmp = scal[3];
    float v = 0.f;
    if (j > i) {
        v = scoreF[idx];
        if (thresh > 3.f && (cf[i] > ftmp) && (cf[j] <= ftmp)) v = 0.f;
    }
    out[idx] = v;
}

// ---------------------------------------------------------------------------
extern "C" void kernel_launch(void* const* d_in, const int* in_sizes, int n_in,
                              void* d_out, int out_size, void* d_ws, size_t ws_size,
                              hipStream_t stream) {
    (void)in_sizes; (void)n_in; (void)out_size; (void)ws_size;
    const float* points = (const float*)d_in[0];   // [384,6]; d_in[1] unused

    char* ws = (char*)d_ws;
    // workspace layout (bytes)
    _Float16* h16A   = (_Float16*)(ws + 0);              // 294912 (f16 score, then f16 T)
    float*    score0 = (float*)(ws + 294912);            // 589824
    float*    scoreF = (float*)(ws + 884736);            // 589824 (sc2, then scoreF in place)
    float*    Tf     = (float*)(ws + 1474560);           // 589824
    float*    T2     = (float*)(ws + 2064384);           // 589824
    float*    cf     = (float*)(ws + 2654208);           // 1536
    float*    cfs    = (float*)(ws + 2655744);           // 1536
    float*    scal   = (float*)(ws + 2657280);           // scalars

    const int nElem = MM;
    const int eBlocks = (nElem + 255) / 256;

    // K1: score (f32 + f16)
    score_kernel<<<eBlocks, 256, 0, stream>>>(points, score0, h16A);
    // K2: sc2 = score @ score  (WMMA, 16x64 strips per wave)
    wmma_symsq<<<36, 128, 0, stream>>>(h16A, scoreF);
    // K3: scoreF = score0*sc2 ; T = cbrt(off-diag)
    combine_kernel<<<eBlocks, 256, 0, stream>>>(score0, scoreF, Tf, h16A);
    // K4: T2 = T @ T  (WMMA)
    wmma_symsq<<<36, 128, 0, stream>>>(h16A, T2);
    // K5: row stats + sums
    rowstats_kernel<<<1, 512, 0, stream>>>(Tf, T2, cf, scal);
    // K6: descending sort
    sortdesc_kernel<<<1, M, 0, stream>>>(cf, cfs);
    // K7: OTSU + scalars
    otsu_kernel<<<1, 32, 0, stream>>>(cfs, scal);
    // K8: prune + triu
    finalize_kernel<<<eBlocks, 256, 0, stream>>>(scoreF, cf, scal, d_out ? (float*)d_out : scoreF);
}
